// LstmNet_36266703848065
// MI455X (gfx1250) — matrix operand order repaired and checked by reading, not compiled, not run on hardware
//
#include <hip/hip_runtime.h>
#include <hip/hip_bf16.h>

typedef __bf16 bf16;
typedef __attribute__((ext_vector_type(16))) __bf16 v16bf;
typedef __attribute__((ext_vector_type(8)))  float  v8f;

// Problem constants
constexpr int    Bn   = 64;
constexpr int    Tn   = 1024;
constexpr int    Hh   = 256;
constexpr int    DIN  = 136;
constexpr int    KP   = 160;        // DIN padded to multiple of 32
constexpr int    G    = 1024;       // 4*H gate width
constexpr size_t Mrows = (size_t)Bn * Tn;   // 65536

// Workspace layout (bytes)
constexpr size_t SZ_XH   = Mrows * KP * sizeof(bf16);          // x hi  (20.97 MB)
constexpr size_t SZ_XL   = SZ_XH;                              // x lo
constexpr size_t SZ_WIH  = 2ull * G * KP * sizeof(bf16);       // w_ih hi (both dirs)
constexpr size_t SZ_WHH  = 2ull * G * Hh * sizeof(bf16);       // w_hh hi (both dirs)
constexpr size_t SZ_PROJ = 2ull * Mrows * G * sizeof(float);   // 536.9 MB
constexpr size_t SZ_HS   = 2ull * Bn * Tn * Hh * sizeof(float);// 134.2 MB

constexpr size_t OFF_XH   = 0;
constexpr size_t OFF_XL   = OFF_XH  + SZ_XH;
constexpr size_t OFF_WIHH = OFF_XL  + SZ_XL;
constexpr size_t OFF_WIHL = OFF_WIHH + SZ_WIH;
constexpr size_t OFF_WHHH = OFF_WIHL + SZ_WIH;
constexpr size_t OFF_WHHL = OFF_WHHH + SZ_WHH;
constexpr size_t OFF_PROJ = OFF_WHHL + SZ_WHH;
constexpr size_t OFF_HS   = OFF_PROJ + SZ_PROJ;

union V16U { v16bf v; uint4 q[2]; };

// ---- WMMA wrapper: D = A(16x32 bf16) * B(32x16 bf16) + C(16x16 f32) ----
__device__ __forceinline__ v8f wmma_bf16(v16bf a, v16bf b, v8f c) {
  return __builtin_amdgcn_wmma_f32_16x16x32_bf16(
      /*neg_a=*/false, a, /*neg_b=*/false, b,
      /*c_mod=*/(short)0, c, /*reuse_a=*/false, /*reuse_b=*/false);
}

// A fragment (16x32 bf16): lane<16 -> row=lane, K = klo+{0..15};
// lane>=16 -> row=lane-16, K = klo+{16..31}.
__device__ __forceinline__ v16bf load_afrag(const bf16* rowbase, int klo, int lane) {
  int kb = klo + ((lane & 16) ? 16 : 0);
  V16U u;
  u.q[0] = *reinterpret_cast<const uint4*>(rowbase + kb);
  u.q[1] = *reinterpret_cast<const uint4*>(rowbase + kb + 8);
  return u.v;
}

// B fragment (32x16 bf16) from W stored row-major (N rows x K cols), so
// B[k][n] = W[n][k].  lane holds column n = n0+(lane&15), K = klo + (lane<16?0:16)..+15
__device__ __forceinline__ v16bf load_bfrag(const bf16* w, int rowStride, int n,
                                            int klo, int lane) {
  const bf16* p = w + (size_t)n * rowStride + klo + ((lane & 16) ? 16 : 0);
  V16U u;
  u.q[0] = *reinterpret_cast<const uint4*>(p);
  u.q[1] = *reinterpret_cast<const uint4*>(p + 8);
  return u.v;
}

__device__ __forceinline__ float sigm(float x) { return 1.0f / (1.0f + __expf(-x)); }

// ---------------------------------------------------------------------------
// Kernel 0: split f32 -> (hi,lo) bf16, with K padding (zero fill).
// ---------------------------------------------------------------------------
__global__ void split_pad_kernel(const float* __restrict__ src,
                                 bf16* __restrict__ hi, bf16* __restrict__ lo,
                                 int rows, int srcK, int dstK) {
  size_t idx = (size_t)blockIdx.x * blockDim.x + threadIdx.x;
  size_t total = (size_t)rows * dstK;
  if (idx >= total) return;
  int r = (int)(idx / dstK), k = (int)(idx % dstK);
  float v = (k < srcK) ? src[(size_t)r * srcK + k] : 0.0f;
  bf16 h = (bf16)v;
  hi[idx] = h;
  lo[idx] = (bf16)(v - (float)h);
}

// ---------------------------------------------------------------------------
// Kernel 1: input projection  proj[d][r][n] = x[r] . w_ih[d][n] + b_ih[d][n]
// r = b*T + t (natural x row order).  Split-bf16 3-product for ~fp32 accuracy.
// grid = (M/128, G/64, 2), block = 256 (8 waves; wave w owns M-tile w).
// ---------------------------------------------------------------------------
__global__ __launch_bounds__(256)
void proj_gemm_kernel(const bf16* __restrict__ xhi, const bf16* __restrict__ xlo,
                      const bf16* __restrict__ wih_hi, const bf16* __restrict__ wih_lo,
                      const float* __restrict__ bih_f, const float* __restrict__ bih_b,
                      float* __restrict__ proj) {
  const int dir  = blockIdx.z;
  const int wave = threadIdx.x >> 5;
  const int lane = threadIdx.x & 31;
  const int m0   = blockIdx.x * 128 + wave * 16;
  const int n0   = blockIdx.y * 64;
  const int nc   = lane & 15;
  const int mofs = (lane & 16) ? 8 : 0;

  const bf16* whi = wih_hi + (size_t)dir * G * KP;
  const bf16* wlo = wih_lo + (size_t)dir * G * KP;
  const float* bih = dir ? bih_b : bih_f;

  v8f acc[4];
  for (int j = 0; j < 4; ++j) {
    float bv = bih[n0 + j * 16 + nc];
    for (int r = 0; r < 8; ++r) acc[j][r] = bv;
  }

  const bf16* xh_row = xhi + (size_t)(m0 + nc) * KP;
  const bf16* xl_row = xlo + (size_t)(m0 + nc) * KP;

  for (int klo = 0; klo < KP; klo += 32) {
    v16bf ah = load_afrag(xh_row, klo, lane);
    v16bf al = load_afrag(xl_row, klo, lane);
    for (int j = 0; j < 4; ++j) {
      int n = n0 + j * 16 + nc;
      v16bf bh = load_bfrag(whi, KP, n, klo, lane);
      v16bf bl = load_bfrag(wlo, KP, n, klo, lane);
      acc[j] = wmma_bf16(ah, bh, acc[j]);   // hi*hi
      acc[j] = wmma_bf16(al, bh, acc[j]);   // lo*hi
      acc[j] = wmma_bf16(ah, bl, acc[j]);   // hi*lo
    }
  }

  float* pbase = proj + (size_t)dir * Mrows * G;
  for (int j = 0; j < 4; ++j)
    for (int r = 0; r < 8; ++r)
      pbase[(size_t)(m0 + mofs + r) * G + (n0 + j * 16 + nc)] = acc[j][r];
}

// ---------------------------------------------------------------------------
// Kernel 2: persistent LSTM scan.  One block per (direction, 16-row batch tile);
// h,c live in LDS; per-step gates = proj[t] + b_hh + h @ w_hh^T via WMMA.
// grid = 8 blocks, block = 256 (8 waves; wave w owns gate columns [128w,128w+128)).
// ---------------------------------------------------------------------------
__global__ __launch_bounds__(256)
void lstm_scan_kernel(const float* __restrict__ proj,
                      const bf16* __restrict__ whh_hi, const bf16* __restrict__ whh_lo,
                      const float* __restrict__ bhh_f, const float* __restrict__ bhh_b,
                      float* __restrict__ hs) {
  const int dir   = blockIdx.x >> 2;
  const int btile = blockIdx.x & 3;
  const int bbase = btile * 16;
  const int wave  = threadIdx.x >> 5;
  const int lane  = threadIdx.x & 31;
  const int nc    = lane & 15;
  const int mofs  = (lane & 16) ? 8 : 0;
  const int n0w   = wave * 128;

  __shared__ alignas(16) bf16  sh_hi[16 * 256];
  __shared__ alignas(16) bf16  sh_lo[16 * 256];
  __shared__ alignas(16) float sc   [16 * 256];
  __shared__ alignas(16) float sg   [16 * 1024];

  for (int i = threadIdx.x; i < 16 * 256; i += 256) {
    sh_hi[i] = (bf16)0.0f; sh_lo[i] = (bf16)0.0f; sc[i] = 0.0f;
  }
  __syncthreads();

  const bf16*  whi  = whh_hi + (size_t)dir * G * Hh;
  const bf16*  wlo  = whh_lo + (size_t)dir * G * Hh;
  const float* bhh  = dir ? bhh_b : bhh_f;
  const float* pdir = proj + (size_t)dir * Mrows * G;
  float*       hdir = hs   + (size_t)dir * Bn * Tn * Hh;

  for (int step = 0; step < Tn; ++step) {
    const int t = dir ? (Tn - 1 - step) : step;

    // ---- gates = proj[t] + b_hh + h @ w_hh^T  (split-bf16, f32 accum) ----
    v8f acc[8];
    for (int j = 0; j < 8; ++j) {
      const int n = n0w + j * 16 + nc;
      const float bv = bhh[n];
      for (int r = 0; r < 8; ++r)
        acc[j][r] = pdir[((size_t)(bbase + mofs + r) * Tn + t) * G + n] + bv;
    }

    // prefetch next step's proj strip into L2
    if (step + 1 < Tn) {
      const int tn = dir ? (t - 1) : (t + 1);
      __builtin_prefetch(&pdir[((size_t)(bbase + nc) * Tn + tn) * G + n0w], 0, 1);
    }

    const bf16* hh_row = sh_hi + nc * 256;
    const bf16* hl_row = sh_lo + nc * 256;
    for (int klo = 0; klo < Hh; klo += 32) {
      v16bf ah = load_afrag(hh_row, klo, lane);
      v16bf al = load_afrag(hl_row, klo, lane);
      for (int j = 0; j < 8; ++j) {
        const int n = n0w + j * 16 + nc;
        v16bf bh = load_bfrag(whi, Hh, n, klo, lane);
        v16bf bl = load_bfrag(wlo, Hh, n, klo, lane);
        acc[j] = wmma_bf16(ah, bh, acc[j]);
        acc[j] = wmma_bf16(al, bh, acc[j]);
        acc[j] = wmma_bf16(ah, bl, acc[j]);
      }
    }

    for (int j = 0; j < 8; ++j)
      for (int r = 0; r < 8; ++r)
        sg[(mofs + r) * 1024 + n0w + j * 16 + nc] = acc[j][r];
    __syncthreads();

    // ---- elementwise gate math (PyTorch order i,f,g,o), update c,h ----
    for (int cell = threadIdx.x; cell < 16 * 256; cell += 256) {
      const int m = cell >> 8, hc = cell & 255;
      const float gi = sg[m * 1024 +   0 + hc];
      const float gf = sg[m * 1024 + 256 + hc];
      const float gg = sg[m * 1024 + 512 + hc];
      const float go = sg[m * 1024 + 768 + hc];
      const float iv = sigm(gi);
      const float fv = sigm(gf);
      const float gv = tanhf(gg);
      const float ov = sigm(go);
      const float cv = fv * sc[cell] + iv * gv;
      sc[cell] = cv;
      const float hv = ov * tanhf(cv);
      const bf16 hb = (bf16)hv;
      sh_hi[cell] = hb;
      sh_lo[cell] = (bf16)(hv - (float)hb);
      hdir[((size_t)(bbase + m) * Tn + t) * Hh + hc] = hv;
    }
    __syncthreads();
  }
}

// ---------------------------------------------------------------------------
// Kernel 3: final FC.  One wave per (b,t): out[b,t,:] = [h_f|h_b] @ fc_w^T + fc_b
// ---------------------------------------------------------------------------
__global__ __launch_bounds__(256)
void fc_kernel(const float* __restrict__ hs, const float* __restrict__ fcw,
               const float* __restrict__ fcb, float* __restrict__ out) {
  const int wave = threadIdx.x >> 5;
  const int lane = threadIdx.x & 31;
  const size_t idx = (size_t)blockIdx.x * 8 + wave;   // b*T + t
  const int b = (int)(idx >> 10);
  const int t = (int)(idx & 1023);

  float acc[5] = {0.f, 0.f, 0.f, 0.f, 0.f};
  for (int f = lane; f < 512; f += 32) {
    const float v = (f < 256)
        ? hs[((size_t)b * Tn + t) * Hh + f]
        : hs[((size_t)(Bn + b) * Tn + t) * Hh + (f - 256)];
    for (int j = 0; j < 5; ++j) acc[j] += v * fcw[j * 512 + f];
  }
  for (int off = 16; off > 0; off >>= 1)
    for (int j = 0; j < 5; ++j) acc[j] += __shfl_xor(acc[j], off, 32);
  if (lane == 0)
    for (int j = 0; j < 5; ++j) out[idx * 5 + j] = acc[j] + fcb[j];
}

// ---------------------------------------------------------------------------
extern "C" void kernel_launch(void* const* d_in, const int* in_sizes, int n_in,
                              void* d_out, int out_size, void* d_ws, size_t ws_size,
                              hipStream_t stream) {
  const float* x      = (const float*)d_in[0];
  const float* w_ih_f = (const float*)d_in[1];
  const float* w_hh_f = (const float*)d_in[2];
  const float* b_ih_f = (const float*)d_in[3];
  const float* b_hh_f = (const float*)d_in[4];
  const float* w_ih_b = (const float*)d_in[5];
  const float* w_hh_b = (const float*)d_in[6];
  const float* b_ih_b = (const float*)d_in[7];
  const float* b_hh_b = (const float*)d_in[8];
  const float* fc_w   = (const float*)d_in[9];
  const float* fc_b   = (const float*)d_in[10];
  float* out = (float*)d_out;

  char* ws = (char*)d_ws;
  bf16*  xhi    = (bf16*)(ws + OFF_XH);
  bf16*  xlo    = (bf16*)(ws + OFF_XL);
  bf16*  wih_hi = (bf16*)(ws + OFF_WIHH);
  bf16*  wih_lo = (bf16*)(ws + OFF_WIHL);
  bf16*  whh_hi = (bf16*)(ws + OFF_WHHH);
  bf16*  whh_lo = (bf16*)(ws + OFF_WHHL);
  float* proj   = (float*)(ws + OFF_PROJ);
  float* hs     = (float*)(ws + OFF_HS);

  // 0) split/pad into bf16 hi/lo
  {
    size_t tot = Mrows * KP;
    split_pad_kernel<<<(unsigned)((tot + 255) / 256), 256, 0, stream>>>(
        x, xhi, xlo, (int)Mrows, DIN, KP);
    size_t totw = (size_t)G * KP;
    split_pad_kernel<<<(unsigned)((totw + 255) / 256), 256, 0, stream>>>(
        w_ih_f, wih_hi, wih_lo, G, DIN, KP);
    split_pad_kernel<<<(unsigned)((totw + 255) / 256), 256, 0, stream>>>(
        w_ih_b, wih_hi + (size_t)G * KP, wih_lo + (size_t)G * KP, G, DIN, KP);
    size_t toth = (size_t)G * Hh;
    split_pad_kernel<<<(unsigned)((toth + 255) / 256), 256, 0, stream>>>(
        w_hh_f, whh_hi, whh_lo, G, Hh, Hh);
    split_pad_kernel<<<(unsigned)((toth + 255) / 256), 256, 0, stream>>>(
        w_hh_b, whh_hi + (size_t)G * Hh, whh_lo + (size_t)G * Hh, G, Hh, Hh);
  }

  // 1) input projections (both directions), WMMA split-bf16
  {
    dim3 grid((unsigned)(Mrows / 128), G / 64, 2);
    proj_gemm_kernel<<<grid, 256, 0, stream>>>(xhi, xlo, wih_hi, wih_lo,
                                               b_ih_f, b_ih_b, proj);
  }

  // 2) recurrent scan (8 persistent blocks: 2 dirs x 4 batch tiles)
  lstm_scan_kernel<<<8, 256, 0, stream>>>(proj, whh_hi, whh_lo, b_hh_f, b_hh_b, hs);

  // 3) final FC
  fc_kernel<<<(unsigned)(Mrows / 8), 256, 0, stream>>>(hs, fc_w, fc_b, out);
}
